// model_283467841597
// MI455X (gfx1250) — matrix operand order repaired and checked by reading, not compiled, not run on hardware
//
#include <hip/hip_runtime.h>

#define TT   10
#define HD   1024
#define VV   50257
#define VPAD 50272      // ceil(V/16)*16 = 3142*16
#define G4H  4096

typedef __bf16 bf16;
typedef __attribute__((ext_vector_type(16))) __bf16 v16bf;
typedef __attribute__((ext_vector_type(8)))  __bf16 v8bf;
typedef __attribute__((ext_vector_type(8)))  float  v8f;

__device__ __forceinline__ bf16 f2bf(float f) { return (bf16)f; }

// ---------------------------------------------------------------------------
// Core: C(16x16,f32) += W[m0:m0+16, 0:K](bf16, row-major) x xsh(bf16, LDS,
// broadcast into all 16 B columns) via v_wmma_f32_16x16x32_bf16.
// ISA layouts (CDNA5 ISA):
//   A (16x32 bf16): lane L<16 -> M=L, K=8*hi+0..7 and 16+8*hi+0..7
//   B (32x16 bf16): lane holds column N=lane&15, K = 8*hi+0..7 and 16+8*hi+0..7
//   C/D (16x16 f32): c[r] -> row M = r + 8*hi, col N = lane&15
// ---------------------------------------------------------------------------
__device__ __forceinline__ void wmma_mv_accum(v8f& c, const bf16* __restrict__ W,
                                              const bf16* xsh,  // LDS
                                              int K, int m0, int lane) {
    const int hi = lane >> 4;
    const bf16* row = W + (size_t)(m0 + (lane & 15)) * (size_t)K;
    for (int k0 = 0; k0 < K; k0 += 32) {
        v8bf alo = *(const v8bf*)(row + k0 + 8 * hi);
        v8bf ahi = *(const v8bf*)(row + k0 + 16 + 8 * hi);
        v8bf blo = *(const v8bf*)(xsh + k0 + 8 * hi);
        v8bf bhi = *(const v8bf*)(xsh + k0 + 16 + 8 * hi);
        v16bf a, b;
#pragma unroll
        for (int i = 0; i < 8; ++i) {
            a[i] = alo[i]; a[8 + i] = ahi[i];
            b[i] = blo[i]; b[8 + i] = bhi[i];
        }
        if (k0 + 32 < K) __builtin_prefetch(row + k0 + 32, 0, 3);  // global_prefetch_b8
        c = __builtin_amdgcn_wmma_f32_16x16x32_bf16(false, a, false, b,
                                                    (short)0, c, false, false);
    }
}

// ---------------------------------------------------------------------------
// gates = W1@x1 + W2@x2 + b1 + b2 + addv   (pointers may be null; x* are bf16)
// Block: 128 threads = 4 waves; activation vectors staged to LDS once.
// ---------------------------------------------------------------------------
__global__ __launch_bounds__(128)
void lstm_gates_wmma(const bf16* __restrict__ W1, const bf16* __restrict__ x1, int K1,
                     const bf16* __restrict__ W2, const bf16* __restrict__ x2, int K2,
                     const float* __restrict__ b1, const float* __restrict__ b2,
                     const float* __restrict__ addv,
                     float* __restrict__ out, int rows) {
    __shared__ __align__(16) bf16 xsh[3 * HD];            // up to 2H + H bf16 = 6KB
    const int tid = (int)threadIdx.x;
    const int total = (K1 + K2) >> 3;                     // 16B chunks
    for (int ch = tid; ch < total; ch += (int)blockDim.x) {
        const int e = ch << 3;
        v8bf v = (e < K1) ? *(const v8bf*)(x1 + e) : *(const v8bf*)(x2 + (e - K1));
        *(v8bf*)(xsh + e) = v;                            // ds_store_b128
    }
    __syncthreads();

    const int wave = (int)((blockIdx.x * blockDim.x + threadIdx.x) >> 5);
    const int lane = tid & 31;
    const int m0 = wave * 16;
    if (m0 >= rows) return;                               // wave-uniform (after barrier)
    v8f c = {};
    if (W1) wmma_mv_accum(c, W1, xsh, K1, m0, lane);
    if (W2) wmma_mv_accum(c, W2, xsh + K1, K2, m0, lane);
    if ((lane & 15) == 0) {
        const int hi = lane >> 4;
#pragma unroll
        for (int r = 0; r < 8; ++r) {
            const int row = m0 + 8 * hi + r;
            float v = c[r];
            if (b1)   v += b1[row];
            if (b2)   v += b2[row];
            if (addv) v += addv[row];
            out[row] = v;
        }
    }
}

// ---------------------------------------------------------------------------
// Batched encoder input GEMM: Gih[t][m] = enc_w_ih @ x_t + b_ih for t=0..T-1
// xseq_bf is a zero-padded 16-column bf16 panel -> branch-free B columns.
// ---------------------------------------------------------------------------
__global__ __launch_bounds__(128)
void enc_input_gemm(const bf16* __restrict__ Wih, const bf16* __restrict__ xseq_bf,
                    const float* __restrict__ bih, float* __restrict__ Gih) {
    __shared__ __align__(16) bf16 xsh[16 * HD];           // 32KB of 320KB WGP LDS
    const int tid = (int)threadIdx.x;
    for (int ch = tid; ch < (16 * HD) >> 3; ch += (int)blockDim.x) {
        const int e = ch << 3;
        *(v8bf*)(xsh + e) = *(const v8bf*)(xseq_bf + e);
    }
    __syncthreads();

    const int wave = (int)((blockIdx.x * blockDim.x + threadIdx.x) >> 5);
    const int lane = tid & 31;
    const int m0 = wave * 16;
    if (m0 >= G4H) return;
    const int hi  = lane >> 4;
    const int col = lane & 15;
    const bf16* row = Wih + (size_t)(m0 + col) * (size_t)HD;
    const bf16* xs  = xsh + col * HD;
    v8f c = {};
    for (int k0 = 0; k0 < HD; k0 += 32) {
        v8bf alo = *(const v8bf*)(row + k0 + 8 * hi);
        v8bf ahi = *(const v8bf*)(row + k0 + 16 + 8 * hi);
        v8bf blo = *(const v8bf*)(xs + k0 + 8 * hi);
        v8bf bhi = *(const v8bf*)(xs + k0 + 16 + 8 * hi);
        v16bf a, b;
#pragma unroll
        for (int i = 0; i < 8; ++i) {
            a[i] = alo[i]; a[8 + i] = ahi[i];
            b[i] = blo[i]; b[8 + i] = bhi[i];
        }
        if (k0 + 32 < HD) __builtin_prefetch(row + k0 + 32, 0, 3);
        c = __builtin_amdgcn_wmma_f32_16x16x32_bf16(false, a, false, b,
                                                    (short)0, c, false, false);
    }
    if (col < TT) {
#pragma unroll
        for (int r = 0; r < 8; ++r) {
            const int rw = m0 + 8 * hi + r;
            Gih[(size_t)col * G4H + rw] = c[r] + bih[rw];
        }
    }
}

// ---------------------------------------------------------------------------
// Vocab projection + fused argmax (103MB bf16 stream, L2-resident):
// preds = out_w @ h + out_b ; key = monotonic(logit)<<32 | ~row -> first-max.
// ---------------------------------------------------------------------------
__global__ __launch_bounds__(128)
void wmma_logits(const bf16* __restrict__ Wbf, const bf16* __restrict__ h_bf,
                 const float* __restrict__ bias, float* __restrict__ preds,
                 unsigned long long* __restrict__ slot) {
    __shared__ __align__(16) bf16 xsh[HD];
    const int tid = (int)threadIdx.x;
    {   // 1024 bf16 = 128 chunks of 16B, one per thread
        const int e = tid << 3;
        *(v8bf*)(xsh + e) = *(const v8bf*)(h_bf + e);
    }
    __syncthreads();

    const int wave = (int)((blockIdx.x * blockDim.x + threadIdx.x) >> 5);
    const int lane = tid & 31;
    const int m0 = wave * 16;
    if (m0 >= VPAD) return;
    v8f c = {};
    wmma_mv_accum(c, Wbf, xsh, HD, m0, lane);
    if ((lane & 15) == 0) {
        const int hi = lane >> 4;
        unsigned long long best = 0ull;
#pragma unroll
        for (int r = 0; r < 8; ++r) {
            const int row = m0 + 8 * hi + r;
            if (row < VV) {
                const float v = c[r] + bias[row];
                preds[row] = v;
                unsigned u = __float_as_uint(v);
                u = (u & 0x80000000u) ? ~u : (u | 0x80000000u);
                const unsigned long long key =
                    ((unsigned long long)u << 32) | (unsigned)(0xFFFFFFFFu - (unsigned)row);
                best = best > key ? best : key;
            }
        }
        if (best) atomicMax(slot, best);
    }
}

// --------------------------- element-wise kernels ---------------------------

__global__ void cvt_bf16(const float* __restrict__ src, bf16* __restrict__ dst, long n) {
    for (long i = (long)blockIdx.x * blockDim.x + threadIdx.x; i < n;
         i += (long)gridDim.x * blockDim.x)
        dst[i] = f2bf(src[i]);
}

__global__ void cvt_bf16_pad_rows(const float* __restrict__ src, bf16* __restrict__ dst,
                                  int validRows, int padRows, int K) {
    const long n = (long)padRows * K;
    for (long i = (long)blockIdx.x * blockDim.x + threadIdx.x; i < n;
         i += (long)gridDim.x * blockDim.x) {
        const long row = i / K;
        dst[i] = (row < validRows) ? f2bf(src[i]) : f2bf(0.0f);
    }
}

__global__ void cvt_bf16_transpose(const float* __restrict__ src, bf16* __restrict__ dst, int n) {
    const long total = (long)n * n;
    for (long i = (long)blockIdx.x * blockDim.x + threadIdx.x; i < total;
         i += (long)gridDim.x * blockDim.x) {
        const long r = i / n, col = i % n;              // dst = src^T
        dst[i] = f2bf(src[col * (long)n + r]);
    }
}

// xseq_bf: 16 columns (t=0..9 real, 10..15 zero) of H bf16 each
__global__ void gather_embed(const int* __restrict__ seq, const float* __restrict__ embeds,
                             bf16* __restrict__ xseq_bf) {
    const int i = (int)(blockIdx.x * blockDim.x + threadIdx.x);
    if (i < 16 * HD) {
        const int t = i >> 10, k = i & (HD - 1);
        const float v = (t < TT) ? embeds[(size_t)seq[t] * HD + k] : 0.0f;
        xseq_bf[i] = f2bf(v);
    }
}

__global__ void init_state(float* h, float* c, float* prev, bf16* h_bf) {
    const int i = (int)threadIdx.x;
    h[i] = 0.f; c[i] = 0.f; prev[i] = 0.f; h_bf[i] = f2bf(0.f);
}

__global__ __launch_bounds__(HD)
void lstm_cell(const float* __restrict__ g, float* __restrict__ h, float* __restrict__ c,
               bf16* __restrict__ h_bf, float* __restrict__ out_h,
               unsigned long long* __restrict__ slot) {
    const int j = (int)threadIdx.x;
    const float ig = 1.f / (1.f + expf(-g[j]));
    const float fg = 1.f / (1.f + expf(-g[j + HD]));
    const float gg = tanhf(g[j + 2 * HD]);
    const float og = 1.f / (1.f + expf(-g[j + 3 * HD]));
    const float cn = fg * c[j] + ig * gg;
    c[j] = cn;
    const float hn = og * tanhf(cn);
    h[j] = hn;
    h_bf[j] = f2bf(hn);                             // bf16 activation for next WMMA
    if (out_h) out_h[j] = hn;
    if (slot && j == 0) *slot = 0ull;               // reset argmax before logits
}

__global__ __launch_bounds__(HD)
void attention_ctx(const float* __restrict__ u, const float* __restrict__ enc_out,
                   const float* __restrict__ prev, bf16* __restrict__ rnn_in_bf) {
    __shared__ float s[TT];
    __shared__ float aw[TT];
    const int tid = (int)threadIdx.x, wave = tid >> 5, lane = tid & 31;
    if (wave < TT) {
        float sum = 0.f;
        const float* e = enc_out + (size_t)wave * HD;
        for (int k = lane; k < HD; k += 32) sum += u[k] * e[k];
        for (int o = 16; o > 0; o >>= 1) sum += __shfl_down(sum, o, 32);
        if (lane == 0) s[wave] = sum;
    }
    __syncthreads();
    if (tid == 0) {
        float mx = s[0];
        for (int j = 1; j < TT; ++j) mx = fmaxf(mx, s[j]);
        float den = 0.f;
        for (int j = 0; j < TT; ++j) { aw[j] = expf(s[j] - mx); den += aw[j]; }
        const float inv = 1.f / den;
        for (int j = 0; j < TT; ++j) aw[j] *= inv;
    }
    __syncthreads();
    float ctx = 0.f;
    for (int j = 0; j < TT; ++j) ctx += aw[j] * enc_out[(size_t)j * HD + tid];
    rnn_in_bf[tid]      = f2bf(fmaxf(prev[tid], 0.f));   // relu(prev)
    rnn_in_bf[HD + tid] = f2bf(ctx);
}

__global__ __launch_bounds__(HD)
void greedy_select(const unsigned long long* __restrict__ slot,
                   const float* __restrict__ embeds, float* __restrict__ prev,
                   float* __restrict__ predseq, int t) {
    const unsigned long long key = *slot;
    const unsigned idx = 0xFFFFFFFFu - (unsigned)(key & 0xFFFFFFFFu);
    const int tid = (int)threadIdx.x;
    prev[tid] = embeds[(size_t)idx * HD + tid];
    if (tid == 0) predseq[t] = (float)idx;
}

// ---------------------------------------------------------------------------
extern "C" void kernel_launch(void* const* d_in, const int* in_sizes, int n_in,
                              void* d_out, int out_size, void* d_ws, size_t ws_size,
                              hipStream_t stream) {
    (void)in_sizes; (void)n_in; (void)out_size; (void)ws_size;
    const int*   seq      = (const int*)  d_in[0];
    const float* embeds   = (const float*)d_in[1];
    const float* enc_w_ih = (const float*)d_in[2];
    const float* enc_w_hh = (const float*)d_in[3];
    const float* enc_b_ih = (const float*)d_in[4];
    const float* enc_b_hh = (const float*)d_in[5];
    const float* dec_w_ih = (const float*)d_in[6];
    const float* dec_w_hh = (const float*)d_in[7];
    const float* dec_b_ih = (const float*)d_in[8];
    const float* dec_b_hh = (const float*)d_in[9];
    const float* out_w    = (const float*)d_in[10];
    const float* out_b    = (const float*)d_in[11];
    const float* wa       = (const float*)d_in[12];

    float* preds   = (float*)d_out;            // (T, V)
    float* predseq = preds + (size_t)TT * VV;  // (T,)

    // ---- workspace layout (256B aligned slices) ----
    char* ws = (char*)d_ws;
    size_t off = 0;
    auto bump = [&](size_t bytes) -> size_t {
        size_t p = (off + 255) & ~(size_t)255; off = p + bytes; return p;
    };
    float* enc_out = (float*)(ws + bump(sizeof(float) * TT * HD));
    float* Gih     = (float*)(ws + bump(sizeof(float) * TT * G4H));
    float* h       = (float*)(ws + bump(sizeof(float) * HD));
    float* cst     = (float*)(ws + bump(sizeof(float) * HD));
    float* prev    = (float*)(ws + bump(sizeof(float) * HD));
    float* uvec    = (float*)(ws + bump(sizeof(float) * HD));
    float* gates   = (float*)(ws + bump(sizeof(float) * G4H));
    unsigned long long* slot = (unsigned long long*)(ws + bump(sizeof(unsigned long long)));
    bf16* xseq_bf    = (bf16*)(ws + bump((size_t)2 * 16 * HD));
    bf16* h_bf       = (bf16*)(ws + bump((size_t)2 * HD));
    bf16* rnn_in_bf  = (bf16*)(ws + bump((size_t)2 * 2 * HD));
    bf16* enc_wih_bf = (bf16*)(ws + bump((size_t)2 * G4H * HD));
    bf16* enc_whh_bf = (bf16*)(ws + bump((size_t)2 * G4H * HD));
    bf16* dec_wih_bf = (bf16*)(ws + bump((size_t)2 * G4H * 2 * HD));
    bf16* dec_whh_bf = (bf16*)(ws + bump((size_t)2 * G4H * HD));
    bf16* waT_bf     = (bf16*)(ws + bump((size_t)2 * HD * HD));
    bf16* outw_bf    = (bf16*)(ws + bump((size_t)2 * VPAD * HD));

    const dim3 cvtGrid(2048), cvtBlk(256);
    // bf16 weight staging: out_w becomes 103MB (fits 192MB L2) for its 10 passes
    cvt_bf16<<<cvtGrid, cvtBlk, 0, stream>>>(enc_w_ih, enc_wih_bf, (long)G4H * HD);
    cvt_bf16<<<cvtGrid, cvtBlk, 0, stream>>>(enc_w_hh, enc_whh_bf, (long)G4H * HD);
    cvt_bf16<<<cvtGrid, cvtBlk, 0, stream>>>(dec_w_ih, dec_wih_bf, (long)G4H * 2 * HD);
    cvt_bf16<<<cvtGrid, cvtBlk, 0, stream>>>(dec_w_hh, dec_whh_bf, (long)G4H * HD);
    cvt_bf16_transpose<<<cvtGrid, cvtBlk, 0, stream>>>(wa, waT_bf, HD);
    cvt_bf16_pad_rows<<<cvtGrid, cvtBlk, 0, stream>>>(out_w, outw_bf, VV, VPAD, HD);

    gather_embed<<<(16 * HD + 255) / 256, 256, 0, stream>>>(seq, embeds, xseq_bf);
    init_state<<<1, HD, 0, stream>>>(h, cst, prev, h_bf);

    // batched encoder input GEMM: Gih[t] = enc_w_ih @ x_t + b_ih (10 real B columns)
    enc_input_gemm<<<G4H / 16 / 4, 128, 0, stream>>>(enc_wih_bf, xseq_bf, enc_b_ih, Gih);

    const int blkGate  = G4H / 16 / 4;         // 256 tiles, 4 waves/block
    const int blkH     = HD  / 16 / 4;         // 64 tiles
    const int blkLogit = (VPAD / 16 + 3) / 4;  // 3142 tiles

    // ---- encoder scan ----
    for (int t = 0; t < TT; ++t) {
        lstm_gates_wmma<<<blkGate, 128, 0, stream>>>(
            nullptr, nullptr, 0, enc_whh_bf, h_bf, HD,
            nullptr, enc_b_hh, Gih + (size_t)t * G4H, gates, G4H);
        lstm_cell<<<1, HD, 0, stream>>>(gates, h, cst, h_bf,
                                        enc_out + (size_t)t * HD, nullptr);
    }

    // ---- decoder scan: attention + greedy feedback ----
    for (int t = 0; t < TT; ++t) {
        // u = wa^T @ h
        lstm_gates_wmma<<<blkH, 128, 0, stream>>>(
            waT_bf, h_bf, HD, nullptr, nullptr, 0,
            nullptr, nullptr, nullptr, uvec, HD);
        attention_ctx<<<1, HD, 0, stream>>>(uvec, enc_out, prev, rnn_in_bf);
        lstm_gates_wmma<<<blkGate, 128, 0, stream>>>(
            dec_wih_bf, rnn_in_bf, 2 * HD, dec_whh_bf, h_bf, HD,
            dec_b_ih, dec_b_hh, nullptr, gates, G4H);
        lstm_cell<<<1, HD, 0, stream>>>(gates, h, cst, h_bf, nullptr, slot);
        wmma_logits<<<blkLogit, 128, 0, stream>>>(
            outw_bf, h_bf, out_b, preds + (size_t)t * VV, slot);
        greedy_select<<<1, HD, 0, stream>>>(slot, embeds, prev, predseq, t);
    }
}
